// TimeLSTM_37374805410596
// MI455X (gfx1250) — matrix operand order repaired and checked by reading, not compiled, not run on hardware
//
#include <hip/hip_runtime.h>
#include <hip/hip_bf16.h>

// ---------------------------------------------------------------------------
// TimeLSTM on MI455X (gfx1250), wave32 + V_WMMA_F32_16X16X32_BF16.
//
// B=128, T=512, IN=512, H=512, G=4H=2048, K=IN+H=1024.
// Persistent kernel: 16 WGs x 512 threads (16 waves). WG owns 32 hidden units
// (all 4 gates -> gating is WG-local). Wave (ct, mhalf): col tile ct (16 cols),
// row tiles mhalf*4..+3. Double-buffered fragment loads + sched_group_barrier
// hints overlap L2 latency with WMMA; a memory-clobber asm at the top of the
// t-loop stops LICM from hoisting (and spilling) the t-invariant W fragments
// while preserving the global address space (global_load_b128, LOADcnt only).
// h/c state lives in registers; h is published as bf16 via a double-buffered
// global buffer + one grid barrier per step. Steady-state working set
// (Xb 64MB + Wb 4MB + hb 256KB) is L2-resident (192MB L2); HBM sees only the
// one-time convert pass and the non-temporal output stores. Workspace ~69MB.
// ---------------------------------------------------------------------------

#define B_   128
#define T_   512
#define IN_  512
#define H_   512
#define G_   2048
#define K_   1024
#define NWG  16

typedef __attribute__((ext_vector_type(16))) __bf16 v16bf;
typedef __attribute__((ext_vector_type(8)))  float  v8f;

#if __has_builtin(__builtin_amdgcn_sched_group_barrier)
#define SGB(mask, size) __builtin_amdgcn_sched_group_barrier((mask), (size), 0)
#else
#define SGB(mask, size)
#endif
// masks: 0x20 = VMEM read, 0x8 = MFMA/WMMA

__device__ __forceinline__ unsigned short f2bf(float f) {
  unsigned int u = __float_as_uint(f);
  u += 0x7FFFu + ((u >> 16) & 1u);           // round-to-nearest-even
  return (unsigned short)(u >> 16);
}
__device__ __forceinline__ float bf2f(unsigned short h) {
  return __uint_as_float(((unsigned int)h) << 16);
}
__device__ __forceinline__ float rcp_fast(float x) {
  return __builtin_amdgcn_rcpf(x);
}
__device__ __forceinline__ float sigm_fast(float x) {
  return rcp_fast(1.0f + __expf(-x));        // v_exp_f32 + v_rcp_f32
}
__device__ __forceinline__ float tanh_fast(float x) {
  return 1.0f - 2.0f * rcp_fast(__expf(2.0f * x) + 1.0f);
}

// ---------------- prep: zero h0 + barrier ----------------------------------
__global__ __launch_bounds__(256) void init_kernel(unsigned short* hb0, int* bar) {
  int idx = blockIdx.x * 256 + threadIdx.x;
  if (idx < B_ * H_) hb0[idx] = 0;
  if (idx < 2) bar[idx] = 0;
}

// ---------------- prep: X f32 [B,T,IN] -> bf16 [T,B,IN] --------------------
__global__ __launch_bounds__(256) void convert_X(const float* __restrict__ X,
                                                 unsigned short* __restrict__ Xb) {
  size_t o = ((size_t)blockIdx.x * 256 + threadIdx.x) * 8;   // out-linear
  int k  = (int)(o & (IN_ - 1));
  size_t bt = o >> 9;
  int b = (int)(bt & (B_ - 1));
  int t = (int)(bt >> 7);
  const float* src = X + ((size_t)b * T_ + t) * IN_ + k;
  union { unsigned short u[8]; uint4 q; } pk;
#pragma unroll
  for (int e = 0; e < 8; ++e) pk.u[e] = f2bf(src[e]);
  *(uint4*)(Xb + o) = pk.q;
}

// ---------------- prep: [Wx;Wh] -> bf16 B-fragment order -------------------
// Fragment (nt, kt): lane l, element e = W[kt*32 + 8*(l>>4) + e + (e&8)][nt*16+(l&15)]
// stored at ((nt*32 + kt)*32 + l)*16 + e.
__global__ __launch_bounds__(256) void convert_W(const float* __restrict__ Wx,
                                                 const float* __restrict__ Wh,
                                                 unsigned short* __restrict__ Wb) {
  int f    = blockIdx.x * 256 + threadIdx.x;    // 131072 fragment-lanes
  int lane = f & 31;
  int kt   = (f >> 5) & 31;
  int nt   = f >> 10;
  int col  = nt * 16 + (lane & 15);
  int kbase = kt * 32 + 8 * (lane >> 4);
  unsigned short* dst = Wb + (size_t)f * 16;
#pragma unroll
  for (int e = 0; e < 16; ++e) {
    int kr = kbase + e + (e & 8);
    float v = (kr < IN_) ? Wx[(size_t)kr * G_ + col]
                         : Wh[(size_t)(kr - IN_) * G_ + col];
    dst[e] = f2bf(v);
  }
}

// ---------------- grid barrier (sense-reversing) ---------------------------
__device__ __forceinline__ void grid_sync(int* cnt, int* gen, int nwg) {
  __threadfence();
  __syncthreads();
  if (threadIdx.x == 0) {
    int g = __hip_atomic_load(gen, __ATOMIC_RELAXED, __HIP_MEMORY_SCOPE_AGENT);
    int p = __hip_atomic_fetch_add(cnt, 1, __ATOMIC_ACQ_REL, __HIP_MEMORY_SCOPE_AGENT);
    if (p == nwg - 1) {
      __hip_atomic_store(cnt, 0, __ATOMIC_RELAXED, __HIP_MEMORY_SCOPE_AGENT);
      __hip_atomic_fetch_add(gen, 1, __ATOMIC_ACQ_REL, __HIP_MEMORY_SCOPE_AGENT);
    } else {
      while (__hip_atomic_load(gen, __ATOMIC_ACQUIRE, __HIP_MEMORY_SCOPE_AGENT) == g) {
        __builtin_amdgcn_s_sleep(2);
      }
    }
  }
  __syncthreads();
  __threadfence();
}

// ---------------- fragment helpers -----------------------------------------
struct AF { uint4 q0, q1; };                     // one 16x32 bf16 A fragment/lane
__device__ __forceinline__ v16bf af_v(const AF& a) {
  union { AF s; v16bf v; } u; u.s = a; return u.v;
}
__device__ __forceinline__ const unsigned short*
a_base(int kt, const unsigned short* Xrow, const unsigned short* hb, int khalf) {
  return (kt < 16 ? Xrow + kt * 32 : hb + (kt - 16) * 32) + 8 * khalf;
}
__device__ __forceinline__ void load_a4(const unsigned short* base, int mhalf,
                                        int l15, AF a[4]) {
#pragma unroll
  for (int m = 0; m < 4; ++m) {
    const unsigned short* ap = base + (size_t)((mhalf * 4 + m) * 16 + l15) * 512;
    a[m].q0 = *(const uint4*)(ap);               // K offsets +0..7
    a[m].q1 = *(const uint4*)(ap + 16);          // K offsets +16..23
  }
}
__device__ __forceinline__ v16bf load_b(const unsigned short* Wfrag, int kt, int lane) {
  return *(const v16bf*)(Wfrag + ((size_t)(kt * 32 + lane) << 4));
}

// ---------------- persistent recurrent kernel ------------------------------
__global__ __launch_bounds__(512, 1)
void lstm_kernel(const unsigned short* __restrict__ Xb,
                 const unsigned short* __restrict__ Wb,
                 unsigned short* hb0, unsigned short* hb1,
                 const float* __restrict__ bias,
                 float* __restrict__ out,
                 int* cnt, int* gen) {
  __shared__ unsigned short As[128 * 132];       // bf16 tanh(preact), stride 132

  const int tid   = threadIdx.x;
  const int wg    = blockIdx.x;                  // hidden slice wg*32..+32
  const int wave  = tid >> 5;                    // 0..15
  const int lane  = tid & 31;
  const int l15   = lane & 15;
  const int khalf = lane >> 4;
  const int ct    = wave & 7;                    // column tile within slice
  const int mhalf = wave >> 3;                   // row-tile half: rows mhalf*64..
  const int gidx  = ct >> 1;                     // gate: 0=f 1=g 2=i 3=o
  const int sub   = ct & 1;
  const int nt    = gidx * 32 + wg * 2 + sub;    // global 16-col tile
  const float bias_l = bias[nt * 16 + l15];

  const unsigned short* Wfrag = Wb + (size_t)nt * (32 * 32 * 16);

  // recurrent state in registers: thread owns (b,j) pairs idx = rep*512+tid
  float hreg[8], creg[8];
#pragma unroll
  for (int r = 0; r < 8; ++r) { hreg[r] = 0.0f; creg[r] = 0.0f; }

  for (int t = 0; t < T_; ++t) {
    // memory clobber: W-fragment loads are t-invariant; without this LLVM
    // hoists all 32 of them (256 VGPRs) out of the t-loop and spills to
    // scratch. Clobber keeps them per-iteration while preserving the global
    // address space (global_load_b128, LOADcnt-only waits).
    asm volatile("" ::: "memory");

    const unsigned short* hb_cur = (t & 1) ? hb1 : hb0;
    unsigned short*       hb_nxt = (t & 1) ? hb0 : hb1;
    const unsigned short* Xrow = Xb + (size_t)t * B_ * IN_;

    // ---- GEMM: A-slice = [x_t | h] @ W, software-pipelined over kt ------
    v8f acc[4] = {};
    AF aE[4], aO[4];
    v16bf bE, bO;
    load_a4(a_base(0, Xrow, hb_cur, khalf), mhalf, l15, aE);
    bE = load_b(Wfrag, 0, lane);
    for (int kt = 0; kt < 32; kt += 2) {
      // issue odd-tile loads while even-tile WMMAs run
      load_a4(a_base(kt + 1, Xrow, hb_cur, khalf), mhalf, l15, aO);
      bO = load_b(Wfrag, kt + 1, lane);
#pragma unroll
      for (int m = 0; m < 4; ++m)
        acc[m] = __builtin_amdgcn_wmma_f32_16x16x32_bf16(
            false, af_v(aE[m]), false, bE, (short)0, acc[m], false, false);
      if (kt + 2 < 32) {
        load_a4(a_base(kt + 2, Xrow, hb_cur, khalf), mhalf, l15, aE);
        bE = load_b(Wfrag, kt + 2, lane);
      }
#pragma unroll
      for (int m = 0; m < 4; ++m)
        acc[m] = __builtin_amdgcn_wmma_f32_16x16x32_bf16(
            false, af_v(aO[m]), false, bO, (short)0, acc[m], false, false);
      // pin the pipeline: [10 loads][4 wmma][10 loads][4 wmma] per kt-pair
      SGB(0x20, 10);
      SGB(0x08, 4);
      SGB(0x20, 10);
      SGB(0x08, 4);
    }

    // ---- A = tanh(acc + bias) -> LDS (bf16) -----------------------------
#pragma unroll
    for (int m = 0; m < 4; ++m) {
      int mt = mhalf * 4 + m;
#pragma unroll
      for (int r = 0; r < 8; ++r) {
        int row = mt * 16 + r + 8 * khalf;
        As[row * 132 + ct * 16 + l15] = f2bf(tanh_fast(acc[m][r] + bias_l));
      }
    }
    __syncthreads();

    // ---- gating + state update (h,c in registers) -----------------------
#pragma unroll
    for (int rep = 0; rep < 8; ++rep) {
      int idx = rep * 512 + tid;
      int b = idx >> 5;
      int j = idx & 31;                          // local hidden unit
      float Af = bf2f(As[b * 132 + j]);
      float Ag = bf2f(As[b * 132 + 32 + j]);
      float Ai = bf2f(As[b * 132 + 64 + j]);
      float Ao = bf2f(As[b * 132 + 96 + j]);
      float fg = sigm_fast(Af);
      float gg = tanh_fast(Ag);
      float ig = sigm_fast(Ai);
      float og = sigm_fast(Ao);
      float c_old = creg[rep];
      // reference stores h BEFORE the update at timestep t (NT: write-once)
      __builtin_nontemporal_store(hreg[rep],
          &out[((size_t)b * T_ + t) * H_ + wg * 32 + j]);
      creg[rep] = fg * c_old + gg * ig;
      float h_new = og * tanh_fast(c_old);       // uses OLD cell state
      hreg[rep] = h_new;
      hb_nxt[b * H_ + wg * 32 + j] = f2bf(h_new);
    }

    grid_sync(cnt, gen, NWG);
  }
}

// ---------------------------------------------------------------------------
extern "C" void kernel_launch(void* const* d_in, const int* in_sizes, int n_in,
                              void* d_out, int out_size, void* d_ws, size_t ws_size,
                              hipStream_t stream) {
  const float* X    = (const float*)d_in[0];   // [B,T,IN] f32
  const float* Wx   = (const float*)d_in[1];   // [IN,4H]  f32
  const float* Wh   = (const float*)d_in[2];   // [H,4H]   f32
  const float* bias = (const float*)d_in[3];   // [4H]     f32
  float* out = (float*)d_out;                  // [B,T,H]  f32

  char* ws = (char*)d_ws;
  int* bar = (int*)ws;                                          // 2 ints
  unsigned short* hb0 = (unsigned short*)(ws + 256);            // 128KB
  unsigned short* hb1 = hb0 + (size_t)B_ * H_;                  // 128KB
  unsigned short* Wb  = hb1 + (size_t)B_ * H_;                  // 4MB
  unsigned short* Xb  = Wb  + (size_t)K_ * G_;                  // 64MB

  init_kernel<<<dim3(256), dim3(256), 0, stream>>>(hb0, bar);
  convert_X<<<dim3(16384), dim3(256), 0, stream>>>(X, Xb);      // 33.5M elems
  convert_W<<<dim3(512), dim3(256), 0, stream>>>(Wx, Wh, Wb);   // 131072 frag-lanes
  lstm_kernel<<<dim3(NWG), dim3(512), 0, stream>>>(Xb, Wb, hb0, hb1,
                                                   bias, out, bar, bar + 1);
}
